// MSAAttention_30760555774379
// MI455X (gfx1250) — matrix operand order repaired and checked by reading, not compiled, not run on hardware
//
#include <hip/hip_runtime.h>
#include <hip/hip_bf16.h>
#include <math.h>

#define SS 128
#define RR 256
#define CM 256
#define CZ 128
#define HH 8
#define DD 32
#define HD 256   // H*D

typedef _Float16 f16;
typedef __attribute__((ext_vector_type(16))) _Float16 v16h;
typedef __attribute__((ext_vector_type(8)))  float    v8f;

__device__ __forceinline__ v8f wmma_f16(v16h a, v16h b, v8f c) {
  return __builtin_amdgcn_wmma_f32_16x16x32_f16(false, a, false, b, (short)0, c, false, false);
}

// A operand: 16x32 f16 tile, rows = M. Per ISA: lane m (0-15) half=0: V0..3 K=0..7, V4..7 K=16..23;
// lanes 16-31: K=8..15 / 24..31. Pairs along K are contiguous in row-major memory.
__device__ __forceinline__ v16h load_a_tile(const f16* base, int ld, int row0, int k0) {
  const int lane = threadIdx.x & 31;
  const int m = lane & 15, half = lane >> 4;
  const f16* r = base + (size_t)(row0 + m) * ld + k0;
  v16h a;
#pragma unroll
  for (int p = 0; p < 8; ++p) {
    const int kb = (p < 4) ? (half * 8 + 2 * p) : (16 + half * 8 + 2 * (p - 4));
    a[2 * p]     = r[kb];
    a[2 * p + 1] = r[kb + 1];
  }
  return a;
}

// B operand: 32x16 f16 tile, loaded from K-transposed storage Bt[n][k].
// Lanes 0-15 hold K=0..7,16..23 for column n, lanes 16-31 hold K=8..15,24..31 (same as A).
__device__ __forceinline__ v16h load_b_tile(const f16* baseT, int ld, int col0, int k0) {
  const int lane = threadIdx.x & 31;
  const int n = lane & 15, half = lane >> 4;
  const f16* r = baseT + (size_t)(col0 + n) * ld + k0;
  v16h b;
#pragma unroll
  for (int p = 0; p < 8; ++p) {
    const int kb = (p < 4) ? (half * 8 + 2 * p) : (16 + half * 8 + 2 * (p - 4));
    b[2 * p]     = r[kb];
    b[2 * p + 1] = r[kb + 1];
  }
  return b;
}

// ---------------- K0: weight convert + transpose to f16 ----------------
// All five 256x256 matrices: dst[a*256+b] = (f16) src[b*256+a]
__global__ void k_convert(const float* wq, const float* wk, const float* wv,
                          const float* wg, const float* wo,
                          f16* wqt, f16* wkt, f16* wvt, f16* wgt, f16* wot) {
  int i = blockIdx.x * blockDim.x + threadIdx.x;   // 5*65536 total
  int which = i >> 16;
  int idx = i & 65535;
  int a = idx >> 8, b = idx & 255;
  const float* src = which == 0 ? wq : which == 1 ? wk : which == 2 ? wv : which == 3 ? wg : wo;
  f16* dst = which == 0 ? wqt : which == 1 ? wkt : which == 2 ? wvt : which == 3 ? wgt : wot;
  dst[(size_t)a * 256 + b] = (f16)src[(size_t)b * 256 + a];
}

// ---------------- K1: pair bias zb[h][q][k] = LN(z[q,k,:]) @ wz[:,h] ----------------
__global__ void k_pair_bias(const float* z, const float* g, const float* b,
                            const float* wz, float* zb) {
  int wave = blockIdx.x * (blockDim.x >> 5) + (threadIdx.x >> 5);
  if (wave >= RR * RR) return;
  int lane = threadIdx.x & 31;
  int qi = wave / RR, kj = wave % RR;
  const float* zp = z + ((size_t)qi * RR + kj) * CZ + lane * 4;
  float vals[4];
  float s = 0.f;
#pragma unroll
  for (int i = 0; i < 4; ++i) { vals[i] = zp[i]; s += vals[i]; }
#pragma unroll
  for (int m = 16; m >= 1; m >>= 1) s += __shfl_xor(s, m, 32);
  float mu = s * (1.0f / CZ);
  float v = 0.f;
#pragma unroll
  for (int i = 0; i < 4; ++i) { float d = vals[i] - mu; v += d * d; }
#pragma unroll
  for (int m = 16; m >= 1; m >>= 1) v += __shfl_xor(v, m, 32);
  float rinv = rsqrtf(v * (1.0f / CZ) + 1e-5f);
  float ln[4];
#pragma unroll
  for (int i = 0; i < 4; ++i)
    ln[i] = (vals[i] - mu) * rinv * g[lane * 4 + i] + b[lane * 4 + i];
#pragma unroll
  for (int h = 0; h < HH; ++h) {
    float acc = 0.f;
#pragma unroll
    for (int i = 0; i < 4; ++i) acc += ln[i] * wz[(lane * 4 + i) * HH + h];
#pragma unroll
    for (int m = 16; m >= 1; m >>= 1) acc += __shfl_xor(acc, m, 32);
    if (lane == 0) zb[((size_t)h * RR + qi) * RR + kj] = acc;
  }
}

// ---------------- K2: LN(m) -> f16 mm[s*R + r][c] ----------------
__global__ void k_ln_m(const float* m, const float* g, const float* b, f16* mm) {
  int row = blockIdx.x * (blockDim.x >> 5) + (threadIdx.x >> 5);
  if (row >= SS * RR) return;
  int lane = threadIdx.x & 31;
  const float* mp = m + (size_t)row * CM + lane * 8;
  float vals[8];
  float s = 0.f;
#pragma unroll
  for (int i = 0; i < 8; ++i) { vals[i] = mp[i]; s += vals[i]; }
#pragma unroll
  for (int mk = 16; mk >= 1; mk >>= 1) s += __shfl_xor(s, mk, 32);
  float mu = s * (1.0f / CM);
  float v = 0.f;
#pragma unroll
  for (int i = 0; i < 8; ++i) { float d = vals[i] - mu; v += d * d; }
#pragma unroll
  for (int mk = 16; mk >= 1; mk >>= 1) v += __shfl_xor(v, mk, 32);
  float rinv = rsqrtf(v * (1.0f / CM) + 1e-5f);
  f16* op = mm + (size_t)row * CM + lane * 8;
#pragma unroll
  for (int i = 0; i < 8; ++i)
    op[i] = (f16)((vals[i] - mu) * rinv * g[lane * 8 + i] + b[lane * 8 + i]);
}

// ---------------- K3: Q/K/V/G projections via WMMA ----------------
// One wave per (s, proj, tile_m, tile_n) 16x16 output tile.
__global__ void k_proj(const f16* mm, const f16* wqt, const f16* wkt,
                       const f16* wvt, const f16* wgt, const float* bg,
                       f16* qb, f16* kb, f16* vt, f16* gb) {
  int gw = blockIdx.x * (blockDim.x >> 5) + (threadIdx.x >> 5);
  int tn = gw & 15; gw >>= 4;
  int tm = gw & 15; gw >>= 4;
  int pj = gw & 3;  int s = gw >> 2;
  const f16* A  = mm + (size_t)s * RR * CM;
  const f16* Bt = pj == 0 ? wqt : pj == 1 ? wkt : pj == 2 ? wvt : wgt;
  v8f acc = {};
#pragma unroll
  for (int kk = 0; kk < CM; kk += 32) {
    v16h a = load_a_tile(A, CM, tm * 16, kk);
    v16h b = load_b_tile(Bt, CM, tn * 16, kk);
    acc = wmma_f16(a, b, acc);
  }
  const int lane = threadIdx.x & 31;
  const int col = tn * 16 + (lane & 15);       // h*32+d
  const int h = col >> 5, d = col & 31;
  const int rb = tm * 16 + 8 * (lane >> 4);
#pragma unroll
  for (int p = 0; p < 8; ++p) {
    int r = rb + p;
    float x = acc[p];
    if (pj == 0) {
      qb[(((size_t)s * HH + h) * RR + r) * DD + d] = (f16)(x * 0.17677669529f);
    } else if (pj == 1) {
      kb[(((size_t)s * HH + h) * RR + r) * DD + d] = (f16)x;
    } else if (pj == 2) {
      vt[(((size_t)s * HH + h) * DD + d) * RR + r] = (f16)x;  // transposed [h][d][r]
    } else {
      float gg = 1.0f / (1.0f + expf(-(x + bg[col])));
      gb[((size_t)s * RR + r) * HD + col] = (f16)gg;
    }
  }
}

// ---------------- K4: attention per (s, h, 32-row block); 2 waves/block ----------------
__global__ void k_attn(const f16* qb, const f16* kb, const f16* vt, const f16* gb,
                       const float* zb, const float* mask, f16* ob) {
  __shared__ float sc[2][16 * RR];   // 32 KB biased scores
  __shared__ f16   pb[2][16 * RR];   // 16 KB probabilities
  __shared__ float mb[RR];           // mask bias
  int bid = blockIdx.x;
  int rbk = bid & 7; bid >>= 3;      // 32-row block index
  int h = bid & 7;  int s = bid >> 3;
  const int wid = threadIdx.x >> 5, lane = threadIdx.x & 31;
  for (int i = threadIdx.x; i < RR; i += blockDim.x)
    mb[i] = 1e9f * (mask[(size_t)s * RR + i] - 1.0f);
  __syncthreads();

  const int q0 = rbk * 32 + wid * 16;
  const f16* Q  = qb + (size_t)(s * HH + h) * RR * DD;
  const f16* K  = kb + (size_t)(s * HH + h) * RR * DD;
  const f16* VT = vt + (size_t)(s * HH + h) * DD * RR;
  const float* zbh = zb + (size_t)h * RR * RR;
  float* scw = sc[wid];
  f16*   pbw = pb[wid];

  // scores: one WMMA per 16x16 tile (K-dim = D = 32); K stored [r][d] is already Bt layout
  v16h aq = load_a_tile(Q, DD, q0, 0);
  const int nl = lane & 15, hf = lane >> 4;
#pragma unroll 1
  for (int t = 0; t < 16; ++t) {
    v16h bk = load_b_tile(K, DD, t * 16, 0);
    v8f c = {};
    c = wmma_f16(aq, bk, c);
    int kj = t * 16 + nl;
#pragma unroll
    for (int p = 0; p < 8; ++p) {
      int r = p + 8 * hf;                 // local row
      int qg = q0 + r;
      scw[r * RR + kj] = c[p] + mb[kj] + zbh[(size_t)qg * RR + kj];
    }
  }

  // softmax: two lanes per row, each scans 128 columns
  {
    int row = lane & 15;
    const float* srow = scw + row * RR + hf * 128;
    float mx = -1e30f;
    for (int i = 0; i < 128; ++i) mx = fmaxf(mx, srow[i]);
    mx = fmaxf(mx, __shfl_xor(mx, 16, 32));
    float sm = 0.f;
    for (int i = 0; i < 128; ++i) sm += expf(srow[i] - mx);
    sm += __shfl_xor(sm, 16, 32);
    float inv = 1.0f / sm;
    f16* prow = pbw + row * RR + hf * 128;
    for (int i = 0; i < 128; ++i) prow[i] = (f16)(expf(srow[i] - mx) * inv);
  }

  // O = P @ V : A = probs (from LDS, row-major), B = VT[d][r] (pairs contiguous along r)
  v8f o0 = {}, o1 = {};
#pragma unroll 1
  for (int kk = 0; kk < RR; kk += 32) {
    v16h ap = load_a_tile(pbw, RR, 0, kk);
    v16h b0 = load_b_tile(VT, RR, 0, kk);
    v16h b1 = load_b_tile(VT, RR, 16, kk);
    o0 = wmma_f16(ap, b0, o0);
    o1 = wmma_f16(ap, b1, o1);
  }

  // gate + store o f16 [s*R + r][h*32+d]
#pragma unroll
  for (int p = 0; p < 8; ++p) {
    int qg = q0 + p + 8 * hf;
    size_t base = ((size_t)s * RR + qg) * HD + h * DD;
    float g0 = (float)gb[base + nl];
    float g1 = (float)gb[base + 16 + nl];
    ob[base + nl]      = (f16)(o0[p] * g0);
    ob[base + 16 + nl] = (f16)(o1[p] * g1);
  }
}

// ---------------- K5: output projection out = o @ wo + bo ----------------
__global__ void k_out(const f16* ob, const f16* wot, const float* bo, float* out) {
  int gw = blockIdx.x * (blockDim.x >> 5) + (threadIdx.x >> 5);
  int tn = gw & 15;
  int tm = gw >> 4;                      // 0..2047
  v8f acc = {};
#pragma unroll
  for (int kk = 0; kk < HD; kk += 32) {
    v16h a = load_a_tile(ob, HD, tm * 16, kk);
    v16h b = load_b_tile(wot, HD, tn * 16, kk);
    acc = wmma_f16(a, b, acc);
  }
  const int lane = threadIdx.x & 31;
  const int col = tn * 16 + (lane & 15);
  const float bias = bo[col];
  const int rb = tm * 16 + 8 * (lane >> 4);
#pragma unroll
  for (int p = 0; p < 8; ++p)
    out[(size_t)(rb + p) * CM + col] = acc[p] + bias;
}

extern "C" void kernel_launch(void* const* d_in, const int* in_sizes, int n_in,
                              void* d_out, int out_size, void* d_ws, size_t ws_size,
                              hipStream_t stream) {
  const float* m      = (const float*)d_in[0];
  const float* z      = (const float*)d_in[1];
  const float* mask   = (const float*)d_in[2];
  const float* ln_m_g = (const float*)d_in[3];
  const float* ln_m_b = (const float*)d_in[4];
  const float* ln_z_g = (const float*)d_in[5];
  const float* ln_z_b = (const float*)d_in[6];
  const float* wz     = (const float*)d_in[7];
  const float* wq     = (const float*)d_in[8];
  const float* wk     = (const float*)d_in[9];
  const float* wv     = (const float*)d_in[10];
  const float* wg     = (const float*)d_in[11];
  const float* bg     = (const float*)d_in[12];
  const float* wo     = (const float*)d_in[13];
  const float* bo     = (const float*)d_in[14];
  float* out = (float*)d_out;

  char* ws = (char*)d_ws;
  size_t off = 0;
  float* zb = (float*)(ws + off); off += (size_t)HH * RR * RR * 4;       // 2 MB
  f16* mm   = (f16*)(ws + off);   off += (size_t)SS * RR * CM * 2;       // 16.78 MB
  f16* wqt  = (f16*)(ws + off);   off += (size_t)CM * HD * 2;
  f16* wkt  = (f16*)(ws + off);   off += (size_t)CM * HD * 2;
  f16* wvt  = (f16*)(ws + off);   off += (size_t)CM * HD * 2;
  f16* wgt  = (f16*)(ws + off);   off += (size_t)CM * HD * 2;
  f16* wot  = (f16*)(ws + off);   off += (size_t)CM * HD * 2;
  f16* qb   = (f16*)(ws + off);   off += (size_t)SS * HH * RR * DD * 2;  // 16.78 MB
  f16* kb   = (f16*)(ws + off);   off += (size_t)SS * HH * RR * DD * 2;
  f16* vt   = (f16*)(ws + off);   off += (size_t)SS * HH * DD * RR * 2;
  f16* gb   = (f16*)(ws + off);   off += (size_t)SS * RR * HD * 2;
  f16* ob   = mm;  // mm dead after projections -> reuse for gated attention output

  k_convert  <<<(5 * 65536) / 256, 256, 0, stream>>>(wq, wk, wv, wg, wo,
                                                     wqt, wkt, wvt, wgt, wot);
  k_pair_bias<<<(RR * RR) / 4, 128, 0, stream>>>(z, ln_z_g, ln_z_b, wz, zb);
  k_ln_m     <<<(SS * RR) / 4, 128, 0, stream>>>(m, ln_m_g, ln_m_b, mm);
  k_proj     <<<(SS * 4 * 16 * 16) / 4, 128, 0, stream>>>(mm, wqt, wkt, wvt, wgt, bg,
                                                          qb, kb, vt, gb);
  k_attn     <<<SS * HH * 8, 64, 0, stream>>>(qb, kb, vt, gb, zb, mask, ob);
  k_out      <<<(SS * RR * 16 / 16) / 4 * 16 / 16, 128, 0, stream>>>(ob, wot, bo, out);
  // note: k_out needs SS*RR/16 * 16 waves = 32768 -> 8192 blocks of 4 waves
}
